// AttentionBlock_4252017623148
// MI455X (gfx1250) — matrix operand order repaired and checked by reading, not compiled, not run on hardware
//
#include <hip/hip_runtime.h>
#include <hip/hip_bf16.h>

#define B_    4
#define C_    256
#define N_    4096   // H*W
#define CQK_  32

typedef __attribute__((ext_vector_type(16))) __bf16 v16bf;
typedef __attribute__((ext_vector_type(8)))  float  v8f;
typedef __attribute__((ext_vector_type(4)))  float  f4;

union BFrag { v16bf v; f4 q[2]; };
static_assert(sizeof(BFrag) == 32, "frag size");

// CDNA5 async global->LDS copy of 16 bytes (tracked by ASYNCcnt).
__device__ __forceinline__ void async_ld16(void* lds, const void* g)
{
    unsigned int l32 = (unsigned int)(unsigned long long)lds;
    asm volatile("global_load_async_to_lds_b128 %0, %1, off"
                 :: "v"(l32), "v"(g) : "memory");
}
#define WAIT_ASYNC(n) asm volatile("s_wait_asynccnt " #n ::: "memory")

// Butterfly max within the 16-lane DPP row (our softmax reduction domain)
// using ROW_XMASK (0x160|mask): pure VALU, no LDS round trip.
template<int MASK>
__device__ __forceinline__ float dpp_xmask_max(float v)
{
    int iv = __builtin_bit_cast(int, v);
    int sw = __builtin_amdgcn_update_dpp(iv, iv, 0x160 | MASK, 0xf, 0xf, true);
    return fmaxf(v, __builtin_bit_cast(float, sw));
}

// ---------------------------------------------------------------------------
// Kernel 1: fused 1x1-conv projections, f32 accumulate, bf16 outputs.
//   qT, kT : [B][N][CQK]  (row = pixel, col = channel)
//   vv     : [B][C][N]    (natural)
// ---------------------------------------------------------------------------
__global__ void proj_qkv_bf16(const float* __restrict__ x,
                              const float* __restrict__ Wq, const float* __restrict__ bq,
                              const float* __restrict__ Wk, const float* __restrict__ bk,
                              const float* __restrict__ Wv, const float* __restrict__ bv,
                              __bf16* __restrict__ qT, __bf16* __restrict__ kT,
                              __bf16* __restrict__ vv)
{
    const int n  = blockIdx.x * 64 + threadIdx.x;
    const int og = blockIdx.y;            // 0..39 : 4 q-groups, 4 k-groups, 32 v-groups
    const int b  = blockIdx.z;

    const float* W; const float* bias; int obase;
    if (og < 4)      { W = Wq; bias = bq; obase = og * 8; }
    else if (og < 8) { W = Wk; bias = bk; obase = (og - 4) * 8; }
    else             { W = Wv; bias = bv; obase = (og - 8) * 8; }

    const float* xcol = x + (size_t)b * C_ * N_ + n;
    float acc[8];
#pragma unroll
    for (int r = 0; r < 8; ++r) acc[r] = 0.f;

    for (int c = 0; c < C_; ++c) {
        const float xv = xcol[(size_t)c * N_];
#pragma unroll
        for (int r = 0; r < 8; ++r)
            acc[r] += W[(size_t)(obase + r) * C_ + c] * xv;
    }

    if (og < 8) {
        __bf16* dst = (og < 4 ? qT : kT) + ((size_t)b * N_ + n) * CQK_ + obase;
#pragma unroll
        for (int r = 0; r < 8; ++r) dst[r] = (__bf16)(acc[r] + bias[obase + r]);
    } else {
#pragma unroll
        for (int r = 0; r < 8; ++r)
            vv[((size_t)b * C_ + obase + r) * N_ + n] = (__bf16)(acc[r] + bias[obase + r]);
    }
}

// ---------------------------------------------------------------------------
// Kernel 2: fused flash attention + residual.
// Block = 256 threads = 8 waves, each wave one 16-row tile (same batch).
// 64-column steps: async-DMA next K/V tile (double buffered, 9 async/wave),
// 4 S-WMMAs, DPP-rowmax softmax, P->bf16 LDS relayout, rowsum via ones-WMMA,
// 32 accumulate WMMAs over 256 channels.  Dynamic LDS: 88 KB.
// ---------------------------------------------------------------------------
#define LDSK_OFF   0
#define LDSK_SZ    4096          // 64 rows x 64 B, per buffer
#define LDSV_OFF   (2 * LDSK_SZ)
#define LDSV_SZ    32768         // 256 rows x 128 B, per buffer
#define PBUF_OFF   (LDSV_OFF + 2 * LDSV_SZ)
#define PBUF_SZ    (16 * 64 * 2) // per wave: 16 rows x 64 bf16
#define SMEM_BYTES (PBUF_OFF + 8 * PBUF_SZ)   // 90112 B

__launch_bounds__(256)
__global__ void flash_attn_bf16(const __bf16* __restrict__ qT,
                                const __bf16* __restrict__ kT,
                                const __bf16* __restrict__ vv,
                                const float* __restrict__ x,
                                const float* __restrict__ gamma,
                                float* __restrict__ out)
{
    extern __shared__ __align__(16) char smem[];
    char*   ldsK = smem + LDSK_OFF;
    char*   ldsV = smem + LDSV_OFF;

    const int tid  = threadIdx.x;
    const int lane = tid & 31;
    const int wave = tid >> 5;
    const int hl   = lane >> 4;       // lane half (0/1)
    const int n16  = lane & 15;

    __bf16* Pb = (__bf16*)(smem + PBUF_OFF) + wave * 16 * 64;   // [16][64]

    const int tile  = blockIdx.x * 8 + wave;     // 16-row tile id (8 tiles share batch)
    const int b     = tile / (N_ / 16);
    const int ibase = (tile % (N_ / 16)) * 16;

    const v8f Zero = {0.f,0.f,0.f,0.f,0.f,0.f,0.f,0.f};

    BFrag ones;                       // all-ones B fragment for rowsum-by-WMMA
#pragma unroll
    for (int i = 0; i < 16; ++i) ones.v[i] = (__bf16)1.0f;

    // Q A-fragment: lane M=n16; e -> K=hl*16+e
    BFrag qA;
    {
        const f4* qp = reinterpret_cast<const f4*>(qT + ((size_t)b * N_ + ibase + n16) * CQK_);
        qA.q[0] = qp[2 * hl];
        qA.q[1] = qp[2 * hl + 1];
    }

    float m[8], l[8];
    v8f   O[16];
#pragma unroll
    for (int r = 0; r < 8; ++r) { m[r] = -3.0e38f; l[r] = 0.f; }
#pragma unroll
    for (int ch = 0; ch < 16; ++ch) O[ch] = Zero;

    const char* kgb = (const char*)(kT + (size_t)b * N_ * CQK_);   // [N][32] bf16
    const char* vgb = (const char*)(vv + (size_t)b * C_ * N_);     // [C][N] bf16

    // Cooperative async fill of one 64-col (K,V) tile pair: 9 async insts/wave.
    auto issue_tile = [&](int jb, int buf) {
        // V: 2048 x 16B chunks over 256 threads (8 each); chunk -> c = ch>>3.
#pragma unroll
        for (int q = 0; q < 8; ++q) {
            const int ch = tid + 256 * q;
            async_ld16(ldsV + buf * LDSV_SZ + ch * 16,
                       vgb + (size_t)(ch >> 3) * (N_ * 2) + (size_t)jb * 2 + (ch & 7) * 16);
        }
        // K: rows jb..jb+63 are 4 KB contiguous; 256 x 16B chunks, 1/thread.
        async_ld16(ldsK + buf * LDSK_SZ + tid * 16, kgb + (size_t)jb * 64 + tid * 16);
    };

    constexpr int NSTEP = N_ / 64;   // 64
    issue_tile(0, 0);

    for (int s = 0; s < NSTEP; ++s) {
        const int jb  = s * 64;
        const int cur = s & 1;

        if (s + 1 < NSTEP) {
            issue_tile(jb + 64, cur ^ 1);
            WAIT_ASYNC(9);           // tile s retired (in-order), tile s+1 in flight
        } else {
            WAIT_ASYNC(0);
        }
        __syncthreads();             // all waves' portions of tile s visible

        // 4 S-subtiles: B-frag col=n16, K = hl*16 + 0..15, rows sub*16+n16
        v8f S[4];
#pragma unroll
        for (int sub = 0; sub < 4; ++sub) {
            BFrag kb;
            const f4* kp = (const f4*)(ldsK + cur * LDSK_SZ + (sub * 16 + n16) * 64 + hl * 32);
            kb.q[0] = kp[0]; kb.q[1] = kp[1];
            S[sub] = __builtin_amdgcn_wmma_f32_16x16x32_bf16(false, qA.v, false, kb.v,
                                                             (short)0, Zero, false, false);
        }

        // Online softmax. D layout: VGPR r -> row r+8*hl, col = n16 (+16 per sub).
        float scale[8];
#pragma unroll
        for (int r = 0; r < 8; ++r) {
            float t = fmaxf(fmaxf(S[0][r], S[1][r]), fmaxf(S[2][r], S[3][r]));
            t = dpp_xmask_max<1>(t);
            t = dpp_xmask_max<2>(t);
            t = dpp_xmask_max<4>(t);
            t = dpp_xmask_max<8>(t);             // rowmax across the 16-lane row
            const float mn = fmaxf(m[r], t);
            scale[r] = __expf(m[r] - mn);
            m[r] = mn;
#pragma unroll
            for (int sub = 0; sub < 4; ++sub)
                Pb[(r + 8 * hl) * 64 + n16 + 16 * sub] = (__bf16)__expf(S[sub][r] - mn);
        }

        // same-wave LDS ordering: P stores complete before A-frag loads
        asm volatile("s_wait_dscnt 0" ::: "memory");

        // P A-frags: pa0 = cols 0..31, pa1 = cols 32..63 (row = n16, 128 B rows)
        BFrag pa0, pa1;
        {
            const f4* rp = (const f4*)(Pb + n16 * 64);
            pa0.q[0] = rp[2 * hl];     pa0.q[1] = rp[2 * hl + 1];
            pa1.q[0] = rp[4 + 2 * hl]; pa1.q[1] = rp[5 + 2 * hl];
        }

        // Row sums via ones-WMMA (replicated across lanes in D layout).
        v8f rs = __builtin_amdgcn_wmma_f32_16x16x32_bf16(false, pa0.v, false, ones.v,
                                                         (short)0, Zero, false, false);
        rs = __builtin_amdgcn_wmma_f32_16x16x32_bf16(false, pa1.v, false, ones.v,
                                                     (short)0, rs, false, false);
#pragma unroll
        for (int r = 0; r < 8; ++r) l[r] = l[r] * scale[r] + rs[r];

        // O accumulate over 256 channels (16 chunks), V B-frags from LDS.
#pragma unroll
        for (int ch = 0; ch < 16; ++ch) {
#pragma unroll
            for (int r = 0; r < 8; ++r) O[ch][r] *= scale[r];
            const char* vrow = ldsV + cur * LDSV_SZ + (ch * 16 + n16) * 128;
            BFrag vb0, vb1;   // j-offsets hl*16+0..15 and 32+hl*16+0..15
            const f4* vp0 = (const f4*)(vrow + hl * 32);
            vb0.q[0] = vp0[0]; vb0.q[1] = vp0[1];
            const f4* vp1 = (const f4*)(vrow + 64 + hl * 32);
            vb1.q[0] = vp1[0]; vb1.q[1] = vp1[1];
            O[ch] = __builtin_amdgcn_wmma_f32_16x16x32_bf16(false, pa0.v, false, vb0.v,
                                                            (short)0, O[ch], false, false);
            O[ch] = __builtin_amdgcn_wmma_f32_16x16x32_bf16(false, pa1.v, false, vb1.v,
                                                            (short)0, O[ch], false, false);
        }

        __syncthreads();             // everyone done with tile s before it is overwritten
    }

    // Epilogue: out = gamma * (O / l) + x
    const float g = gamma[0];
    float invl[8];
#pragma unroll
    for (int r = 0; r < 8; ++r) invl[r] = 1.0f / l[r];

    const size_t xb = (size_t)b * C_ * N_;
#pragma unroll
    for (int ch = 0; ch < 16; ++ch) {
        const int c = ch * 16 + n16;
#pragma unroll
        for (int r = 0; r < 8; ++r) {
            const int    i   = ibase + r + 8 * hl;
            const size_t idx = xb + (size_t)c * N_ + i;
            out[idx] = g * (O[ch][r] * invl[r]) + x[idx];
        }
    }
}

// ---------------------------------------------------------------------------
extern "C" void kernel_launch(void* const* d_in, const int* in_sizes, int n_in,
                              void* d_out, int out_size, void* d_ws, size_t ws_size,
                              hipStream_t stream)
{
    const float* x     = (const float*)d_in[0];
    const float* Wq    = (const float*)d_in[1];
    const float* bq    = (const float*)d_in[2];
    const float* Wk    = (const float*)d_in[3];
    const float* bk    = (const float*)d_in[4];
    const float* Wv    = (const float*)d_in[5];
    const float* bv    = (const float*)d_in[6];
    const float* gamma = (const float*)d_in[7];

    char* ws = (char*)d_ws;
    const size_t qk_bytes = (size_t)B_ * N_ * CQK_ * sizeof(__bf16);   // 1 MB each
    __bf16* qT = (__bf16*)(ws);
    __bf16* kT = (__bf16*)(ws + qk_bytes);
    __bf16* vv = (__bf16*)(ws + 2 * qk_bytes);                         // 8 MB

    dim3 pgrid(N_ / 64, 40, B_);
    proj_qkv_bf16<<<pgrid, 64, 0, stream>>>(x, Wq, bq, Wk, bk, Wv, bv, qT, kT, vv);

    dim3 fgrid((B_ * (N_ / 16)) / 8);   // 8 waves (tiles) per 256-thread block
    flash_attn_bf16<<<fgrid, 256, SMEM_BYTES, stream>>>(qT, kT, vv, x, gamma,
                                                        (float*)d_out);
}
